// AttentionBlock_38225208934579
// MI455X (gfx1250) — matrix operand order repaired and checked by reading, not compiled, not run on hardware
//
#include <hip/hip_runtime.h>
#include <hip/hip_bf16.h>

// ---------------------------------------------------------------------------
// Fused graph-attention block for MI455X (gfx1250, wave32, WMMA).
// One workgroup (256 threads = 8 wave32) per graph of 128 atoms x 64 channels.
// - x tile staged via global_load_async_to_lds_b128 (ASYNCcnt path)
// - all contractions on v_wmma_f32_16x16x32_f16 (f16 operands, f32 accum)
// - fast-math norm_act (raw v_sqrt/v_rcp/v_exp/v_log)
// - fully parallel 256-thread column softmax
// ---------------------------------------------------------------------------

typedef __attribute__((ext_vector_type(16))) _Float16     v16h;
typedef __attribute__((ext_vector_type(8)))  float        v8f;
typedef __attribute__((ext_vector_type(4)))  unsigned int u32x4;

#define FDIM   64
#define NATOMS 128
#define NGRAPH 64
#define LN2    0.6931471805599453f

// ---- LDS layout (bytes); zero static LDS so dynamic LDS starts at offset 0,
// making these byte offsets valid raw LDS addresses for async-to-LDS ops.
#define SM_WGT   0                                  // 4*64*64 f16 = 32768
#define SM_XH    (SM_WGT + 4 * FDIM * FDIM * 2)     // 128*64 f16  = 16384 (later qW)
#define SM_QH    (SM_XH + NATOMS * FDIM * 2)        // 128*64 f16
#define SM_KH    (SM_QH + NATOMS * FDIM * 2)        // 128*64 f16
#define SM_VF    (SM_KH + NATOMS * FDIM * 2)        // 128*64 f32 (async x stage, then v)
#define SM_S     (SM_VF + NATOMS * FDIM * 4)        // 128*64 f32 score block
#define SM_SA    (SM_S + NATOMS * FDIM * 4)         // 64 f32
#define SM_RED   (SM_SA + FDIM * 4)                 // 256 f32 softmax reduction
#define SM_TOTAL (SM_RED + 256 * 4)                 // 148,736 B

// A fragment (16x32 f16): lanes 0-15 row m0+l, K quarters [k0..k0+7],[k0+16..23];
// lanes 16-31 row m0+l-16, K [k0+8..15],[k0+24..31].
__device__ __forceinline__ v16h load_frag_a(const _Float16* src, int ld,
                                            int m0, int k0, int lane) {
  const int m  = m0 + (lane & 15);
  const int kb = k0 + ((lane >> 4) << 3);
  union { v16h v; u32x4 u[2]; } t;
  t.u[0] = *reinterpret_cast<const u32x4*>(src + m * ld + kb);
  t.u[1] = *reinterpret_cast<const u32x4*>(src + m * ld + kb + 16);
  return t.v;
}

// B fragment (32x16 f16) from row-major B^T[n][k]: lanes 0-15 col n0+l K [k0..k0+7],[k0+16..23];
// lanes 16-31 col n0+l-16 K [k0+8..15],[k0+24..31].
__device__ __forceinline__ v16h load_frag_b(const _Float16* srcT, int ld,
                                            int n0, int k0, int lane) {
  const int n  = n0 + (lane & 15);
  const int kb = k0 + ((lane >> 4) << 3);
  union { v16h v; u32x4 u[2]; } t;
  t.u[0] = *reinterpret_cast<const u32x4*>(srcT + n * ld + kb);
  t.u[1] = *reinterpret_cast<const u32x4*>(srcT + n * ld + kb + 16);
  return t.v;
}

__device__ __forceinline__ float fast_rcp(float x) {
  float r = __builtin_amdgcn_rcpf(x);
  return r * (2.0f - x * r);                 // one Newton step
}

// NormActivation gate on scalars: (softplus(n)-ln2)/n * y, n = sqrt(y^2+eps^2).
// n >= eps > 0, exp(-n) in (0,1] -> __logf(1+t) is safe & fast (v_exp/v_log).
__device__ __forceinline__ float norm_act(float y) {
  float n  = __builtin_amdgcn_sqrtf(__builtin_fmaf(y, y, 1e-16f));
  float sp = n + __logf(1.0f + __expf(-n));
  return (sp - LN2) * y * fast_rcp(n);
}

__global__ __launch_bounds__(256, 1)
void attn_graph_kernel(const float* __restrict__ x,
                       const float* __restrict__ Wq,
                       const float* __restrict__ Wk,
                       const float* __restrict__ Wv,
                       const float* __restrict__ Wqk,
                       float* __restrict__ out) {
  extern __shared__ char smem[];
  _Float16* wgtT = reinterpret_cast<_Float16*>(smem + SM_WGT);
  _Float16* xh   = reinterpret_cast<_Float16*>(smem + SM_XH);   // later: qW f16
  _Float16* qh   = reinterpret_cast<_Float16*>(smem + SM_QH);
  _Float16* kh   = reinterpret_cast<_Float16*>(smem + SM_KH);
  float*    vf   = reinterpret_cast<float*>(smem + SM_VF);
  float*    S    = reinterpret_cast<float*>(smem + SM_S);
  float*    sa   = reinterpret_cast<float*>(smem + SM_SA);
  float*    red  = reinterpret_cast<float*>(smem + SM_RED);

  const int g    = blockIdx.x;
  const int tid  = threadIdx.x;
  const int lane = tid & 31;
  const int wave = tid >> 5;
  const float* xg = x + (size_t)g * NATOMS * FDIM;

  // ---- async-stage this graph's x tile (f32, 32KB) into the vf buffer ----
  // GVS addressing: mem = SGPR64(xg) + VADDR32(voff); LDS dest = raw byte addr.
  // 2048 x b128 transfers, 8 per thread; overlaps with weight staging below.
#pragma unroll
  for (int i = 0; i < 8; ++i) {
    unsigned idx      = tid + i * 256;          // b128 transfer index
    unsigned lds_addr = SM_VF + idx * 16;
    unsigned voff     = idx * 16;
    asm volatile("global_load_async_to_lds_b128 %0, %1, %2"
                 :: "v"(lds_addr), "v"(voff), "s"(xg) : "memory");
  }

  // ---- stage weights transposed (W^T[n][k]) as f16 (synchronous, overlaps) --
  const float* Ws[4] = {Wq, Wk, Wv, Wqk};
#pragma unroll
  for (int w = 0; w < 4; ++w) {
    const float* W = Ws[w];
    for (int idx = tid; idx < FDIM * FDIM; idx += 256) {
      int kf = idx >> 6, n = idx & 63;
      wgtT[w * FDIM * FDIM + n * FDIM + kf] = (_Float16)W[idx];
    }
  }

  asm volatile("s_wait_asynccnt 0x0" ::: "memory");
  __syncthreads();

  // convert staged x (f32 in vf) -> f16 xh; vf is then free until v is written
  for (int idx = tid; idx < NATOMS * FDIM; idx += 256)
    xh[idx] = (_Float16)vf[idx];
  __syncthreads();

  const int m0     = wave * 16;        // each wave owns one 16-row M tile
  const int rowoff = (lane >> 4) << 3;
  const int nlane  = lane & 15;

  // ---- phase 1 (fused): q,k,v = norm_act(x @ {Wq,Wk,Wv}) ------------------
  // One A-fragment load feeds 12 WMMAs (3 weights x 4 N-tiles).
  {
    v8f acc[3][4];
#pragma unroll
    for (int t = 0; t < 3; ++t)
#pragma unroll
      for (int nt = 0; nt < 4; ++nt) acc[t][nt] = v8f(0.f);
#pragma unroll
    for (int kk = 0; kk < FDIM; kk += 32) {
      v16h a = load_frag_a(xh, FDIM, m0, kk, lane);
#pragma unroll
      for (int t = 0; t < 3; ++t)
#pragma unroll
        for (int nt = 0; nt < 4; ++nt) {
          v16h b = load_frag_b(wgtT + t * FDIM * FDIM, FDIM, nt * 16, kk, lane);
          acc[t][nt] = __builtin_amdgcn_wmma_f32_16x16x32_f16(
              false, a, false, b, (short)0, acc[t][nt], false, false);
        }
    }
#pragma unroll
    for (int t = 0; t < 3; ++t)
#pragma unroll
      for (int nt = 0; nt < 4; ++nt)
#pragma unroll
        for (int r = 0; r < 8; ++r) {
          float y   = norm_act(acc[t][nt][r]);
          int   row = m0 + r + rowoff;
          int   col = nt * 16 + nlane;
          if (t == 0)      qh[row * FDIM + col] = (_Float16)y;
          else if (t == 1) kh[row * FDIM + col] = (_Float16)y;
          else             vf[row * FDIM + col] = y;
        }
  }
  __syncthreads();

  // ---- phase 1b: qW = q @ Wqk, f16 into the (dead) xh buffer --------------
  {
    v8f acc[4] = {v8f(0.f), v8f(0.f), v8f(0.f), v8f(0.f)};
#pragma unroll
    for (int kk = 0; kk < FDIM; kk += 32) {
      v16h a = load_frag_a(qh, FDIM, m0, kk, lane);
#pragma unroll
      for (int nt = 0; nt < 4; ++nt) {
        v16h b = load_frag_b(wgtT + 3 * FDIM * FDIM, FDIM, nt * 16, kk, lane);
        acc[nt] = __builtin_amdgcn_wmma_f32_16x16x32_f16(
            false, a, false, b, (short)0, acc[nt], false, false);
      }
    }
#pragma unroll
    for (int nt = 0; nt < 4; ++nt)
#pragma unroll
      for (int r = 0; r < 8; ++r)
        xh[(m0 + r + rowoff) * FDIM + nt * 16 + nlane] = (_Float16)acc[nt][r];
  }
  __syncthreads();
  const _Float16* qwh = xh;

  // ---- phase 2: scores + column softmax + output, 64 dst columns/block ----
  const int c  = tid & 63;        // softmax column
  const int qt = tid >> 6;        // row slab: qt*32 .. qt*32+31
  for (int jb = 0; jb < NATOMS; jb += 64) {
    // S[i][c] = dot(qW[i], k[jb+c]) / 8   (B operand = kh rows => k^T)
    v8f acc[4] = {v8f(0.f), v8f(0.f), v8f(0.f), v8f(0.f)};
#pragma unroll
    for (int kk = 0; kk < FDIM; kk += 32) {
      v16h a = load_frag_a(qwh, FDIM, m0, kk, lane);
#pragma unroll
      for (int nt = 0; nt < 4; ++nt) {
        v16h b = load_frag_b(kh, FDIM, jb + nt * 16, kk, lane);
        acc[nt] = __builtin_amdgcn_wmma_f32_16x16x32_f16(
            false, a, false, b, (short)0, acc[nt], false, false);
      }
    }
#pragma unroll
    for (int nt = 0; nt < 4; ++nt)
#pragma unroll
      for (int r = 0; r < 8; ++r)
        S[(m0 + r + rowoff) * 64 + nt * 16 + nlane] = acc[nt][r] * 0.125f;
    __syncthreads();

    // parallel column softmax: 4 threads per column, 32-row slabs each,
    // lane-consecutive columns => conflict-free LDS sweeps.
    float pm = -__builtin_inff();
#pragma unroll 8
    for (int i = qt * 32; i < qt * 32 + 32; ++i) pm = fmaxf(pm, S[i * 64 + c]);
    red[qt * 64 + c] = pm;
    __syncthreads();
    float m = fmaxf(fmaxf(red[c], red[64 + c]), fmaxf(red[128 + c], red[192 + c]));
    float ps = 0.f;
#pragma unroll 8
    for (int i = qt * 32; i < qt * 32 + 32; ++i) {
      float e = __expf(S[i * 64 + c] - m);
      S[i * 64 + c] = e;
      ps += e;
    }
    __syncthreads();                 // reads of red done before overwrite
    red[qt * 64 + c] = ps;
    __syncthreads();
    float inv = fast_rcp(red[c] + red[64 + c] + red[128 + c] + red[192 + c]);
    float pa = 0.f;
#pragma unroll 8
    for (int i = qt * 32; i < qt * 32 + 32; ++i) pa += S[i * 64 + c] * inv;
    __syncthreads();
    red[qt * 64 + c] = pa;
    __syncthreads();
    if (tid < 64) sa[c] = red[c] + red[64 + c] + red[128 + c] + red[192 + c];
    __syncthreads();

    // out[j] = x[j] + v[j] * sum_alpha[j]   (reference gathers v[dst])
    for (int idx = tid; idx < 64 * FDIM; idx += 256) {
      int cc = idx >> 6, f = idx & 63;
      int jl = jb + cc;
      out[((size_t)g * NATOMS + jl) * FDIM + f] =
          xg[jl * FDIM + f] + vf[jl * FDIM + f] * sa[cc];
    }
    __syncthreads();
  }
}

extern "C" void kernel_launch(void* const* d_in, const int* in_sizes, int n_in,
                              void* d_out, int out_size, void* d_ws, size_t ws_size,
                              hipStream_t stream) {
  (void)in_sizes; (void)n_in; (void)d_ws; (void)ws_size; (void)out_size;
  const float* x   = (const float*)d_in[0];
  const float* Wq  = (const float*)d_in[1];
  const float* Wk  = (const float*)d_in[2];
  const float* Wv  = (const float*)d_in[3];
  const float* Wqk = (const float*)d_in[4];
  float* out = (float*)d_out;

  hipFuncSetAttribute((const void*)attn_graph_kernel,
                      hipFuncAttributeMaxDynamicSharedMemorySize, SM_TOTAL);
  attn_graph_kernel<<<NGRAPH, 256, SM_TOTAL, stream>>>(x, Wq, Wk, Wv, Wqk, out);
}
